// Net_16681652977771
// MI455X (gfx1250) — matrix operand order repaired and checked by reading, not compiled, not run on hardware
//
#include <hip/hip_runtime.h>
#include <hip/hip_bf16.h>

// ---------------------------------------------------------------------------
// Types / WMMA helpers (CDNA5 gfx1250, wave32)
// ---------------------------------------------------------------------------
typedef __bf16 bf16_t;
typedef __attribute__((ext_vector_type(16))) __bf16 v16bf;
typedef __attribute__((ext_vector_type(8)))  __bf16 v8bf;
typedef __attribute__((ext_vector_type(8)))  float  v8f;

#define B_SZ 4096

// Output tuple offsets (flat, float32):
//   z_logit (B,10) | z_hat (B,1) | mu (B,2) | logvar (B,2) | u_out (B,2) |
//   x_hat_logit (B,1,64,64) | transfo_d1 (B,32) | transfo_d2 (B,32)
#define O_ZL 0
#define O_ZH 40960
#define O_MU 45056
#define O_LV 53248
#define O_UO 61440
#define O_XH 69632
#define O_T1 16846848
#define O_T2 16977920

__device__ __forceinline__ v8f wmma_bf16(v16bf a, v16bf b, v8f c) {
  // D = A(16x32) * B(32x16) + C, fp32 accumulate
  return __builtin_amdgcn_wmma_f32_16x16x32_bf16(false, a, false, b, (short)0, c,
                                                 false, false);
}

// A-matrix (16x32 bf16) fragment: lane = h*16 + r (r = row M).
// elems 0..7  -> K = k0..k0+7       (k0 = kt*32 + h*8)
// elems 8..15 -> K = k0+16..k0+23
__device__ __forceinline__ v16bf frag_a(const bf16_t* __restrict__ row, int k0) {
  union { v8bf h[2]; v16bf f; } u;
  u.h[0] = *reinterpret_cast<const v8bf*>(row + k0);
  u.h[1] = *reinterpret_cast<const v8bf*>(row + k0 + 16);
  return u.f;
}

// B-matrix (32x16 bf16) fragment: lane column n = lane&15, same K order as A:
// elems 0..7 -> K = kb..kb+7, elems 8..15 -> K = kb+16..kb+23.
// row points at W[n][*] (W stored N x K row-major); kb = kt*32 + h*8.
__device__ __forceinline__ v16bf frag_b(const bf16_t* __restrict__ row, int kb) {
  union { v8bf h[2]; v16bf f; } u;
  u.h[0] = *reinterpret_cast<const v8bf*>(row + kb);
  u.h[1] = *reinterpret_cast<const v8bf*>(row + kb + 16);
  return u.f;
}

// One wave computes a 16(M) x 64(N) tile; A-fragment reused across 4 WMMAs.
__device__ __forceinline__ void gemm_core_16x64(const bf16_t* __restrict__ Arow,
                                                const bf16_t* __restrict__ Wrow,
                                                int ldw, int ksteps, int h,
                                                v8f acc[4]) {
#pragma unroll 4
  for (int kt = 0; kt < ksteps; ++kt) {
    const int ka = kt * 32 + h * 8;
    const int kb = kt * 32 + h * 8;
    v16bf a  = frag_a(Arow, ka);
    v16bf b0 = frag_b(Wrow,                    kb);
    v16bf b1 = frag_b(Wrow + (size_t)16 * ldw, kb);
    v16bf b2 = frag_b(Wrow + (size_t)32 * ldw, kb);
    v16bf b3 = frag_b(Wrow + (size_t)48 * ldw, kb);
    acc[0] = wmma_bf16(a, b0, acc[0]);
    acc[1] = wmma_bf16(a, b1, acc[1]);
    acc[2] = wmma_bf16(a, b2, acc[2]);
    acc[3] = wmma_bf16(a, b3, acc[3]);
  }
}

// ---------------------------------------------------------------------------
// Weight conversion: fp32 -> bf16, optional transpose, K zero-padded.
// dst[n*Kd + k] = (k < Ks) ? (tr ? src[k*ld + n] : src[n*ld + k]) : 0
// ---------------------------------------------------------------------------
__global__ void k_cvt_w(bf16_t* __restrict__ dst, const float* __restrict__ src,
                        int N, int Kd, int Ks, int tr, int ld) {
  int i = blockIdx.x * blockDim.x + threadIdx.x;
  if (i >= N * Kd) return;
  int n = i / Kd, k = i % Kd;
  float v = 0.f;
  if (k < Ks) v = tr ? src[(size_t)k * ld + n] : src[(size_t)n * ld + k];
  dst[i] = (bf16_t)v;
}

// ---------------------------------------------------------------------------
// Fused conv1+conv2 through LDS. Block = 256 threads (8 waves) = 2 batch rows.
// Phase 1 (conv1): M = 256 rows/b (blk=i2*4+j2, pq), K=16(pad 32), N=16.
//   D-tiles -> LDS patch-major sh1[bb][blk][c][pq] (conv2's K=256 contiguous).
// Phase 2 (conv2): per wave one 16(M=ij) x 16(N=c) tile, K=256 from LDS.
//   Out h2[b][c][pq] (conv3's K=1024 contiguous).
// ---------------------------------------------------------------------------
__global__ void __launch_bounds__(256) k_conv12(
    const float* __restrict__ xin, const bf16_t* __restrict__ W1,
    const float* __restrict__ b1, const bf16_t* __restrict__ W2,
    const float* __restrict__ b2, bf16_t* __restrict__ h2) {
  __shared__ bf16_t sh1[2 * 4096];  // 16 KB
  const int lane = threadIdx.x & 31;
  const int w    = threadIdx.x >> 5;
  const int r = lane & 15, h = lane >> 4;

  {  // ---- phase 1: conv1, 4 tiles per wave ----
    v16bf bfrag = frag_b(W1 + (size_t)r * 32, h * 8);
    const float bn = b1[r];
#pragma unroll
    for (int u = 0; u < 4; ++u) {
      const int tt = w * 4 + u;          // 0..31 tiles (2 b x 16 blk)
      const int bb = tt >> 4, blk = tt & 15;
      const int b  = blockIdx.x * 2 + bb;
      const int y  = (blk >> 2) * 4 + (r >> 2);
      const int xx = (blk & 3)  * 4 + (r & 3);
      // h=0 -> kernel rows p=0,1 ; h=1 -> p=2,3 ; upper half-K zero-padded.
      const float* px =
          xin + ((size_t)b << 12) + (size_t)(4 * y + 2 * h) * 64 + 4 * xx;
      __builtin_prefetch(px + 4096, 0, 1);  // stream hint: next batch row
      float4 f0 = *reinterpret_cast<const float4*>(px);
      float4 f1 = *reinterpret_cast<const float4*>(px + 64);
      v16bf a = {};
      a[0] = (bf16_t)f0.x; a[1] = (bf16_t)f0.y;
      a[2] = (bf16_t)f0.z; a[3] = (bf16_t)f0.w;
      a[4] = (bf16_t)f1.x; a[5] = (bf16_t)f1.y;
      a[6] = (bf16_t)f1.z; a[7] = (bf16_t)f1.w;
      v8f acc = {};
      acc = wmma_bf16(a, bfrag, acc);
#pragma unroll
      for (int v = 0; v < 8; ++v) {
        float val = fmaxf(acc[v] + bn, 0.f);
        sh1[bb * 4096 + blk * 256 + r * 16 + (v + 8 * h)] = (bf16_t)val;
      }
    }
  }
  __syncthreads();
  {  // ---- phase 2: conv2, one 16x16 tile per wave, A from LDS ----
    const int bb = w >> 2;
    const int n0 = (w & 3) << 4;
    const int b  = blockIdx.x * 2 + bb;
    const bf16_t* Arow = &sh1[bb * 4096 + r * 256];
    const bf16_t* Wrow = W2 + (size_t)(n0 + r) * 256;
    const int n = n0 + r;
    const float bn = b2[n];
    v8f acc = {};
#pragma unroll
    for (int kt = 0; kt < 8; ++kt) {
      v16bf a  = frag_a(Arow, kt * 32 + h * 8);
      v16bf bf = frag_b(Wrow, kt * 32 + h * 8);
      acc = wmma_bf16(a, bf, acc);
    }
#pragma unroll
    for (int v = 0; v < 8; ++v) {
      float val = fmaxf(acc[v] + bn, 0.f);
      h2[(size_t)b * 1024 + n * 16 + (v + 8 * h)] = (bf16_t)val;
    }
  }
}

// conv3: M=4096 (b), K=1024 contiguous, N=256. Out hb[b][o].
__global__ void k_conv3(const bf16_t* __restrict__ h2, const bf16_t* __restrict__ W,
                        const float* __restrict__ bias, bf16_t* __restrict__ hb) {
  const int lane = threadIdx.x & 31;
  const int gw   = blockIdx.x * (blockDim.x >> 5) + (threadIdx.x >> 5);
  const int r = lane & 15, h = lane >> 4;
  const int mt = gw >> 2, nt = gw & 3;
  const int m0 = mt << 4, n0 = nt << 6;
  const bf16_t* Arow = h2 + (size_t)(m0 + r) * 1024;
  const bf16_t* Wrow = W + (size_t)(n0 + r) * 1024;
  v8f acc[4] = {};
  gemm_core_16x64(Arow, Wrow, 1024, 32, h, acc);
#pragma unroll
  for (int t = 0; t < 4; ++t) {
    const int n = n0 + t * 16 + r;
    const float bn = bias[n];
#pragma unroll
    for (int v = 0; v < 8; ++v) {
      const int m = m0 + v + 8 * h;
      float val = fmaxf(acc[t][v] + bn, 0.f);
      hb[(size_t)m * 256 + n] = (bf16_t)val;
    }
  }
}

// x_z / x_u fused: M=4096, K=256, N=64 (rows 0..31 = w_xz, 32..63 = w_xu).
__global__ void k_xzxu(const bf16_t* __restrict__ hb, const bf16_t* __restrict__ W,
                       float* __restrict__ xzxu) {
  const int lane = threadIdx.x & 31;
  const int gw   = blockIdx.x * (blockDim.x >> 5) + (threadIdx.x >> 5);
  const int r = lane & 15, h = lane >> 4;
  const int m0 = gw << 4;
  const bf16_t* Arow = hb + (size_t)(m0 + r) * 256;
  const bf16_t* Wrow = W + (size_t)r * 256;
  v8f acc[4] = {};
  gemm_core_16x64(Arow, Wrow, 256, 8, h, acc);
#pragma unroll
  for (int t = 0; t < 4; ++t) {
    const int n = t * 16 + r;
#pragma unroll
    for (int v = 0; v < 8; ++v) {
      const int m = m0 + v + 8 * h;
      xzxu[(size_t)m * 64 + n] = fmaxf(acc[t][v], 0.f);
    }
  }
}

// ---------------------------------------------------------------------------
// Sampling head: z_logit, categorical (Gumbel-argmax, deterministic hash RNG),
// mu/logvar, reparameterized u_out, trans -> transfo_d1/d2.
// ---------------------------------------------------------------------------
__device__ __forceinline__ unsigned hashu(unsigned x) {
  x ^= x >> 16; x *= 0x7feb352du; x ^= x >> 15; x *= 0x846ca68bu; x ^= x >> 16;
  return x;
}
__device__ __forceinline__ float u01(unsigned s) {
  return (hashu(s) >> 8) * (1.f / 16777216.f) + (0.5f / 16777216.f);
}
__device__ __forceinline__ float nrm(unsigned s) {
  float a = u01(s * 2u + 0x1234567u), b = u01(s * 2u + 0x89abcdfu);
  return sqrtf(fmaxf(-2.f * logf(a), 0.f)) * cosf(6.28318530718f * b);
}

__global__ void k_head(const float* __restrict__ xzxu,
                       const float* __restrict__ w_label,
                       const float* __restrict__ w_zd1,
                       const float* __restrict__ w_zd2,
                       const float* __restrict__ w_mu,
                       const float* __restrict__ w_logvar,
                       float* __restrict__ out) {
  int b = blockIdx.x * blockDim.x + threadIdx.x;
  if (b >= B_SZ) return;
  const float* xz = xzxu + (size_t)b * 64;
  const float* xu = xz + 32;
  float logit[10];
#pragma unroll
  for (int j = 0; j < 10; ++j) {
    float s = 0.f;
#pragma unroll
    for (int i = 0; i < 32; ++i) s += xz[i] * w_label[j * 32 + i];
    logit[j] = s;
    out[O_ZL + (size_t)b * 10 + j] = s;
  }
  int best = 0; float bv = -1e30f;
#pragma unroll
  for (int j = 0; j < 10; ++j) {
    float g = -logf(-logf(u01((unsigned)(b * 10 + j) ^ 0x9e3779b9u)));
    float v = logit[j] + g;
    if (v > bv) { bv = v; best = j; }
  }
  out[O_ZH + b] = (float)best;
  float mu0 = 0, mu1 = 0, lv0 = 0, lv1 = 0;
#pragma unroll
  for (int i = 0; i < 32; ++i) {
    float xv = xu[i];
    mu0 += xv * w_mu[i];       mu1 += xv * w_mu[32 + i];
    lv0 += xv * w_logvar[i];   lv1 += xv * w_logvar[32 + i];
  }
  out[O_MU + (size_t)b * 2]     = mu0;  out[O_MU + (size_t)b * 2 + 1] = mu1;
  out[O_LV + (size_t)b * 2]     = lv0;  out[O_LV + (size_t)b * 2 + 1] = lv1;
  float u0 = nrm((unsigned)(b * 2))     * expf(0.5f * lv0) + mu0;
  float u1 = nrm((unsigned)(b * 2 + 1)) * expf(0.5f * lv1) + mu1;
  out[O_UO + (size_t)b * 2]     = u0;   out[O_UO + (size_t)b * 2 + 1] = u1;
#pragma unroll
  for (int i = 0; i < 32; ++i) {
    out[O_T1 + (size_t)b * 32 + i] = u0 + w_zd1[i * 10 + best];
    out[O_T2 + (size_t)b * 32 + i] = u1 + w_zd2[i * 10 + best];
  }
}

// x_hat[b,o] = sum_i relu(t1*tc_w[i,o,0] + t2*tc_w[i,o,1] + tc_b[i,o])
// (relu inside the reduction -> not a GEMM; block = one b, thread = o)
__global__ void k_xhat(const float* __restrict__ out_ro,
                       const float* __restrict__ tc_w,
                       const float* __restrict__ tc_b,
                       bf16_t* __restrict__ xhatb) {
  const int b = blockIdx.x, o = threadIdx.x;
  __shared__ float s1[32], s2[32];
  if (threadIdx.x < 32) {
    s1[threadIdx.x] = out_ro[O_T1 + (size_t)b * 32 + threadIdx.x];
    s2[threadIdx.x] = out_ro[O_T2 + (size_t)b * 32 + threadIdx.x];
  }
  __syncthreads();
  float acc = 0.f;
#pragma unroll
  for (int i = 0; i < 32; ++i) {
    float v = s1[i] * tc_w[i * 512 + o * 2] + s2[i] * tc_w[i * 512 + o * 2 + 1] +
              tc_b[i * 256 + o];
    acc += fmaxf(v, 0.f);
  }
  xhatb[(size_t)b * 256 + o] = (bf16_t)acc;
}

// dec3: M=4096 (b), K=256, N=1024 (o*16+pq). Out d3[b][pq][c] for dec2 rows.
__global__ void k_dec3(const bf16_t* __restrict__ xhatb, const bf16_t* __restrict__ W,
                       const float* __restrict__ bias, bf16_t* __restrict__ d3) {
  const int lane = threadIdx.x & 31;
  const int gw   = blockIdx.x * (blockDim.x >> 5) + (threadIdx.x >> 5);
  const int r = lane & 15, h = lane >> 4;
  const int mt = gw >> 4, nt = gw & 15;
  const int m0 = mt << 4, n0 = nt << 6;
  const bf16_t* Arow = xhatb + (size_t)(m0 + r) * 256;
  const bf16_t* Wrow = W + (size_t)(n0 + r) * 256;
  v8f acc[4] = {};
  gemm_core_16x64(Arow, Wrow, 256, 8, h, acc);
#pragma unroll
  for (int t = 0; t < 4; ++t) {
    const int n = n0 + t * 16 + r;
    const int o = n >> 4;
    const float bn = bias[o];
#pragma unroll
    for (int v = 0; v < 8; ++v) {
      const int m = m0 + v + 8 * h;
      float val = fmaxf(acc[t][v] + bn, 0.f);
      d3[(size_t)m * 1024 + (n & 15) * 64 + o] = (bf16_t)val;
    }
  }
}

// ---------------------------------------------------------------------------
// Fused dec2+dec1 through LDS. Block = 256 threads (8 waves) = 2 batch rows.
// Phase 1 (dec2): per wave one 16(M=ij) x 64(N=o*16+pq) tile, K=64 from d3.
//   D scattered into LDS sd2[bb][Y][X][o] (LDS scatter is cheap; this was the
//   1/16-sector global store pattern before).
// Phase 2 (dec1): 4 tiles/wave, M=(Y,X) rows, K=16(pad 32) contiguous in LDS,
//   N=16 (pq). fp32 output written coalesced (q-contiguous across lanes).
// ---------------------------------------------------------------------------
__global__ void __launch_bounds__(256) k_dec21(
    const bf16_t* __restrict__ d3, const bf16_t* __restrict__ W2,
    const float* __restrict__ b2, const bf16_t* __restrict__ W1,
    const float* __restrict__ b1, float* __restrict__ out) {
  __shared__ bf16_t sd2[2 * 4096];  // 16 KB
  const int lane = threadIdx.x & 31;
  const int w    = threadIdx.x >> 5;
  const int r = lane & 15, h = lane >> 4;

  {  // ---- phase 1: dec2 ----
    const int bb = w >> 2;
    const int n0 = (w & 3) << 6;
    const int b  = blockIdx.x * 2 + bb;
    const int m0 = b << 4;
    const bf16_t* Arow = d3 + (size_t)(m0 + r) * 64;
    const bf16_t* Wrow = W2 + (size_t)(n0 + r) * 64;
    v8f acc[4] = {};
    gemm_core_16x64(Arow, Wrow, 64, 2, h, acc);
#pragma unroll
    for (int t = 0; t < 4; ++t) {
      const int n = n0 + t * 16 + r;
      const int o = n >> 4, p = (n >> 2) & 3, q = n & 3;
      const float bn = b2[o];
#pragma unroll
      for (int v = 0; v < 8; ++v) {
        const int ij = v + 8 * h;           // row within this b
        const int i = ij >> 2, j = ij & 3;
        float val = fmaxf(acc[t][v] + bn, 0.f);
        sd2[bb * 4096 + (4 * i + p) * 256 + (4 * j + q) * 16 + o] = (bf16_t)val;
      }
    }
  }
  __syncthreads();
  {  // ---- phase 2: dec1, 4 tiles per wave ----
    const float bias0 = b1[0];
    v16bf bfrag = frag_b(W1 + (size_t)r * 32, h * 8);
    const int p = r >> 2, q = r & 3;
#pragma unroll
    for (int u = 0; u < 4; ++u) {
      const int tt = w * 4 + u;             // 0..31 tiles (2 b x 16 Y)
      const int bb = tt >> 4, Y = tt & 15;
      const int b  = blockIdx.x * 2 + bb;
      const bf16_t* row = &sd2[bb * 4096 + (Y * 16 + r) * 16];
      v8bf lo = *reinterpret_cast<const v8bf*>(row + h * 8);
      v16bf a = {};
#pragma unroll
      for (int i = 0; i < 8; ++i) a[i] = lo[i];
      v8f acc = {};
      acc = wmma_bf16(a, bfrag, acc);
#pragma unroll
      for (int v = 0; v < 8; ++v) {
        const int X = v + 8 * h;
        out[O_XH + (size_t)b * 4096 + (size_t)(4 * Y + p) * 64 + 4 * X + q] =
            acc[v] + bias0;
      }
    }
  }
}

// ---------------------------------------------------------------------------
// Host launch
// ---------------------------------------------------------------------------
extern "C" void kernel_launch(void* const* d_in, const int* in_sizes, int n_in,
                              void* d_out, int out_size, void* d_ws, size_t ws_size,
                              hipStream_t stream) {
  (void)in_sizes; (void)n_in; (void)out_size; (void)ws_size;
  const float* x        = (const float*)d_in[0];
  const float* conv1_w  = (const float*)d_in[3];
  const float* conv1_b  = (const float*)d_in[4];
  const float* conv2_w  = (const float*)d_in[5];
  const float* conv2_b  = (const float*)d_in[6];
  const float* conv3_w  = (const float*)d_in[7];
  const float* conv3_b  = (const float*)d_in[8];
  const float* w_xz     = (const float*)d_in[9];
  const float* w_label  = (const float*)d_in[10];
  const float* w_zd1    = (const float*)d_in[11];
  const float* w_zd2    = (const float*)d_in[12];
  const float* w_xu     = (const float*)d_in[13];
  const float* w_mu     = (const float*)d_in[14];
  const float* w_logvar = (const float*)d_in[15];
  const float* tc_w     = (const float*)d_in[16];
  const float* tc_b     = (const float*)d_in[17];
  const float* dec3_w   = (const float*)d_in[18];
  const float* dec3_b   = (const float*)d_in[19];
  const float* dec2_w   = (const float*)d_in[20];
  const float* dec2_b   = (const float*)d_in[21];
  const float* dec1_w   = (const float*)d_in[22];
  const float* dec1_b   = (const float*)d_in[23];
  float* out = (float*)d_out;

  char* p = (char*)d_ws;
  auto take = [&](size_t elems, size_t esz) -> void* {
    void* r = (void*)p;
    p += (elems * esz + 255) & ~(size_t)255;
    return r;
  };
  bf16_t* Wc1  = (bf16_t*)take(16 * 32, 2);
  bf16_t* Wc2  = (bf16_t*)take(64 * 256, 2);
  bf16_t* Wc3  = (bf16_t*)take(256 * 1024, 2);
  bf16_t* Wxz  = (bf16_t*)take(64 * 256, 2);
  bf16_t* Wd3  = (bf16_t*)take(1024 * 256, 2);
  bf16_t* Wd2  = (bf16_t*)take(256 * 64, 2);
  bf16_t* Wd1  = (bf16_t*)take(16 * 32, 2);
  bf16_t* h2   = (bf16_t*)take((size_t)B_SZ * 1024, 2);
  bf16_t* hb   = (bf16_t*)take((size_t)B_SZ * 256, 2);
  bf16_t* xhb  = (bf16_t*)take((size_t)B_SZ * 256, 2);
  bf16_t* d3   = (bf16_t*)take((size_t)B_SZ * 1024, 2);
  float*  xzxu = (float*) take((size_t)B_SZ * 64, 4);

  auto cvt = [&](bf16_t* dst, const float* src, int N, int Kd, int Ks, int tr,
                 int ld) {
    int tot = N * Kd;
    k_cvt_w<<<(tot + 255) / 256, 256, 0, stream>>>(dst, src, N, Kd, Ks, tr, ld);
  };
  cvt(Wc1, conv1_w, 16, 32, 16, 0, 16);            // (16,1,4,4) -> 16x32
  cvt(Wc2, conv2_w, 64, 256, 256, 0, 256);         // (64,16,4,4) -> 64x256
  cvt(Wc3, conv3_w, 256, 1024, 1024, 0, 1024);     // (256,64,4,4) -> 256x1024
  cvt(Wxz, w_xz, 32, 256, 256, 0, 256);            // rows 0..31
  cvt(Wxz + 32 * 256, w_xu, 32, 256, 256, 0, 256); // rows 32..63
  cvt(Wd3, dec3_w, 1024, 256, 256, 1, 1024);       // transpose (c-major -> n,k)
  cvt(Wd2, dec2_w, 256, 64, 64, 1, 256);           // transpose
  cvt(Wd1, dec1_w, 16, 32, 16, 1, 16);             // transpose + pad

  k_conv12<<<2048, 256, 0, stream>>>(x, Wc1, conv1_b, Wc2, conv2_b, h2);
  k_conv3 <<<128,  256, 0, stream>>>(h2, Wc3, conv3_b, hb);
  k_xzxu  <<<32,   256, 0, stream>>>(hb, Wxz, xzxu);
  k_head  <<<16,   256, 0, stream>>>(xzxu, w_label, w_zd1, w_zd2, w_mu,
                                     w_logvar, out);
  k_xhat  <<<4096, 256, 0, stream>>>(out, tc_w, tc_b, xhb);
  k_dec3  <<<512,  256, 0, stream>>>(xhb, Wd3, dec3_b, d3);
  k_dec21 <<<2048, 256, 0, stream>>>(d3, Wd2, dec2_b, Wd1, dec1_b, out);
}
